// OneRelClassifier_33998961115715
// MI455X (gfx1250) — matrix operand order repaired and checked by reading, not compiled, not run on hardware
//
#include <hip/hip_runtime.h>
#include <hip/hip_bf16.h>

typedef __bf16 bf16;
typedef __attribute__((ext_vector_type(16))) __bf16 v16bf;
typedef __attribute__((ext_vector_type(8)))  __bf16 v8bf;
typedef __attribute__((ext_vector_type(8)))  float  v8f;
typedef __attribute__((ext_vector_type(4)))  int    v4i;

#define HDIM 768
#define K1   2304     // 3*H
#define NROW 512      // B*S
#define NO   96       // OUT*TAG

// workspace byte offsets (all 256-aligned)
#define O_XBF   0u          // 512*768   bf16 =   786,432 B
#define O_W1T   786432u     // 2304*1536 bf16 = 7,077,888 B
#define O_W2T   7864320u    // 96*2304   bf16 =   442,368 B
#define O_HEAD  8306688u    // 512*2304  f32  = 4,718,592 B
#define O_TAIL  13025280u   // 512*2304  f32  = 4,718,592 B  (end ~17.7 MB)

#if defined(__has_builtin)
#if __has_builtin(__builtin_amdgcn_global_load_async_to_lds_b128) && \
    __has_builtin(__builtin_amdgcn_s_wait_asynccnt)
#define USE_ASYNC 1
#endif
#endif
#ifndef USE_ASYNC
#define USE_ASYNC 0
#endif

#if USE_ASYNC
#define GPTR(p) ((__attribute__((address_space(1))) v4i*)(p))
#define LPTR(p) ((__attribute__((address_space(3))) v4i*)(p))
#endif

union Frag { v16bf v; v8bf h[2]; };

// ---------------- prep: fp32 -> bf16 copy -------------------------------
__global__ void k_cvt_x(const float* __restrict__ x, bf16* __restrict__ xb, int n) {
    int i = blockIdx.x * 256 + threadIdx.x;
    if (i < n) xb[i] = (bf16)x[i];
}

// ---------------- prep: tiled transpose + convert -----------------------
__global__ void k_transpose_cvt(const float* __restrict__ src, bf16* __restrict__ dst,
                                int rows, int cols) {
    __shared__ float tile[32][33];
    int tx = threadIdx.x & 31, ty = threadIdx.x >> 5;
    int c0 = blockIdx.x * 32, r0 = blockIdx.y * 32;
#pragma unroll
    for (int i = 0; i < 4; ++i)
        tile[ty + i * 8][tx] = src[(size_t)(r0 + ty + i * 8) * cols + c0 + tx];
    __syncthreads();
#pragma unroll
    for (int i = 0; i < 4; ++i)
        dst[(size_t)(c0 + ty + i * 8) * rows + r0 + tx] = (bf16)tile[tx][ty + i * 8];
}

// ---------------- kernel 1: head/tail = x @ W1h / x @ W1t ----------------
__global__ void __launch_bounds__(256) k_gemm1(
        const bf16* __restrict__ xb, const bf16* __restrict__ w1t,
        float* __restrict__ headp, float* __restrict__ tailp) {
    int lane  = threadIdx.x & 31;
    int gw    = blockIdx.x * 8 + (threadIdx.x >> 5);
    int ht    = gw / (32 * 36);
    int rem   = gw % (32 * 36);
    int mt    = rem / 36;
    int ng    = rem % 36;
    int lmod  = lane & 15, lhalf = lane >> 4;

    v8f acc[4] = {};
    const bf16* arow  = xb + (size_t)(mt * 16 + lmod) * HDIM;
    const bf16* bbase = w1t + ht * HDIM;

    for (int kc = 0; kc < HDIM; kc += 32) {
        Frag A, Bm[4];
        A.h[0] = *(const v8bf*)(arow + kc + lhalf * 8);
        A.h[1] = *(const v8bf*)(arow + kc + 16 + lhalf * 8);
#pragma unroll
        for (int t = 0; t < 4; ++t) {
            const bf16* brow = bbase + (size_t)(ng * 64 + t * 16 + lmod) * 1536
                             + kc + lhalf * 8;
            Bm[t].h[0] = *(const v8bf*)(brow);
            Bm[t].h[1] = *(const v8bf*)(brow + 16);
        }
#pragma unroll
        for (int t = 0; t < 4; ++t)
            acc[t] = __builtin_amdgcn_wmma_f32_16x16x32_bf16(
                         false, A.v, false, Bm[t].v, (short)0, acc[t], false, false);
    }
    float* dst = ht ? tailp : headp;
#pragma unroll
    for (int t = 0; t < 4; ++t)
#pragma unroll
        for (int v = 0; v < 8; ++v)
            dst[(size_t)(mt * 16 + v + 8 * lhalf) * K1 + ng * 64 + t * 16 + lmod]
                = acc[t][v];
}

// ---------------- kernel 2: fused pair-GEMM ------------------------------
// One block per (b,i). K-chunk 64. Tail chunk register-double-buffered;
// W2 chunk ping-pong in LDS, staged by async global->LDS copy when available.
__global__ void __launch_bounds__(256) k_pair(
        const float* __restrict__ headp, const float* __restrict__ tailp,
        const float* __restrict__ b1,    const bf16*  __restrict__ w2t,
        const float* __restrict__ b2,    float* __restrict__ out) {
    __shared__ float hb[K1];                         //  9,216 B
    __shared__ __align__(16) bf16 aT[128 * 64];      // 16,384 B
    __shared__ __align__(16) bf16 wT[2][96 * 64];    // 24,576 B

    int tid   = threadIdx.x;
    int bi    = blockIdx.x;            // b*128 + i
    int tbase = (bi >> 7) << 7;        // b*128
    int lane  = tid & 31, wv = tid >> 5;
    int lmod  = lane & 15, lhalf = lane >> 4;

    for (int k = tid; k < K1; k += 256)
        hb[k] = headp[(size_t)bi * K1 + k] + b1[k];

    v8f acc[6] = {};

    // A-tile fill: thread covers row j, 32 consecutive k
    int j  = tid >> 1;
    int kb = (tid & 1) << 5;
    const float* trow = tailp + (size_t)(tbase + j) * K1 + kb;
    // W2 staging: threads 0..191 cover row wn, 32 consecutive k (64 B each)
    int wn = tid >> 1;
    int wk = (tid & 1) << 5;
    const bf16* w2row = w2t + (size_t)wn * K1 + wk;

    const bf16* arow = aT + (wv * 16 + lmod) * 64;

    // ---- prologue: stage chunk 0 ----
    float4 tq[8];
    {
        const float4* tp = (const float4*)(trow);
#pragma unroll
        for (int g = 0; g < 8; ++g) tq[g] = tp[g];
    }
#if USE_ASYNC
    if (tid < 192) {
#pragma unroll
        for (int g = 0; g < 4; ++g)
            __builtin_amdgcn_global_load_async_to_lds_b128(
                GPTR(w2row + g * 8), LPTR(&wT[0][wn * 64 + wk] + g * 8), 0, 0);
    }
#else
    v8bf wq[4];
    if (tid < 192) {
        const v8bf* sp = (const v8bf*)(w2row);
#pragma unroll
        for (int g = 0; g < 4; ++g) wq[g] = sp[g];
    }
#endif
    __syncthreads();   // hb visible

    for (int it = 0; it < 36; ++it) {
        int kc  = it * 64;
        int cur = it & 1;
        // ---- compute A-tile from staged tail regs: relu(hb + tail) ----
        {
            v8bf r[4];
#pragma unroll
            for (int g = 0; g < 4; ++g) {
                float4 qa = tq[2 * g], qb = tq[2 * g + 1];
                float f[8] = {qa.x, qa.y, qa.z, qa.w, qb.x, qb.y, qb.z, qb.w};
#pragma unroll
                for (int q = 0; q < 8; ++q) {
                    float s = f[q] + hb[kc + kb + g * 8 + q];
                    r[g][q] = (bf16)fmaxf(s, 0.0f);
                }
            }
            v8bf* dv = (v8bf*)(aT + j * 64 + kb);
#pragma unroll
            for (int g = 0; g < 4; ++g) dv[g] = r[g];
        }
#if !USE_ASYNC
        if (tid < 192) {
            v8bf* dp = (v8bf*)(&wT[cur][wn * 64 + wk]);
#pragma unroll
            for (int g = 0; g < 4; ++g) dp[g] = wq[g];
        }
#endif
        // ---- prefetch next chunk into registers (overlaps WMMA phase) ----
        int kcn = (it < 35) ? kc + 64 : kc;
        {
            const float4* tp = (const float4*)(trow + kcn);
#pragma unroll
            for (int g = 0; g < 8; ++g) tq[g] = tp[g];
        }
#if !USE_ASYNC
        if (tid < 192) {
            const v8bf* sp = (const v8bf*)(w2row + kcn);
#pragma unroll
            for (int g = 0; g < 4; ++g) wq[g] = sp[g];
        }
#endif
#if USE_ASYNC
        __builtin_amdgcn_s_wait_asynccnt(0);   // wT[cur] fully written
#endif
        __syncthreads();   // publish aT (+ wT[cur])
#if USE_ASYNC
        // async-stage next W2 chunk into the other buffer; overlaps WMMAs
        if (tid < 192 && it < 35) {
#pragma unroll
            for (int g = 0; g < 4; ++g)
                __builtin_amdgcn_global_load_async_to_lds_b128(
                    GPTR(w2row + kc + 64 + g * 8),
                    LPTR(&wT[cur ^ 1][wn * 64 + wk] + g * 8), 0, 0);
        }
#endif
        // ---- two 32-wide sub-chunks; batch frags, burst WMMAs ----
        const bf16* wbase = &wT[cur][0];
        Frag A0, A1, Bf[6];
        A0.h[0] = *(const v8bf*)(arow + lhalf * 8);
        A0.h[1] = *(const v8bf*)(arow + 16 + lhalf * 8);
        A1.h[0] = *(const v8bf*)(arow + 32 + lhalf * 8);
        A1.h[1] = *(const v8bf*)(arow + 48 + lhalf * 8);
#pragma unroll
        for (int t = 0; t < 6; ++t) {
            const bf16* brow = wbase + (t * 16 + lmod) * 64 + lhalf * 8;
            Bf[t].h[0] = *(const v8bf*)(brow);
            Bf[t].h[1] = *(const v8bf*)(brow + 16);
        }
#pragma unroll
        for (int t = 0; t < 6; ++t)
            acc[t] = __builtin_amdgcn_wmma_f32_16x16x32_bf16(
                         false, A0.v, false, Bf[t].v, (short)0, acc[t], false, false);
#pragma unroll
        for (int t = 0; t < 6; ++t) {
            const bf16* brow = wbase + (t * 16 + lmod) * 64 + 32 + lhalf * 8;
            Bf[t].h[0] = *(const v8bf*)(brow);
            Bf[t].h[1] = *(const v8bf*)(brow + 16);
        }
#pragma unroll
        for (int t = 0; t < 6; ++t)
            acc[t] = __builtin_amdgcn_wmma_f32_16x16x32_bf16(
                         false, A1.v, false, Bf[t].v, (short)0, acc[t], false, false);
        __syncthreads();   // protect aT overwrite next iteration
    }

    // ---- epilogue: + b2, scatter per C/D lane layout ----
    float* op = out + (size_t)bi * 128 * NO;
#pragma unroll
    for (int t = 0; t < 6; ++t) {
        int o = t * 16 + lmod;
        float bb = b2[o];
#pragma unroll
        for (int v = 0; v < 8; ++v) {
            int jj = wv * 16 + v + 8 * lhalf;
            op[jj * NO + o] = acc[t][v] + bb;
        }
    }
}

// ---------------- launcher ----------------------------------------------
extern "C" void kernel_launch(void* const* d_in, const int* in_sizes, int n_in,
                              void* d_out, int out_size, void* d_ws, size_t ws_size,
                              hipStream_t stream) {
    const float* x  = (const float*)d_in[0];
    const float* W1 = (const float*)d_in[1];
    const float* b1 = (const float*)d_in[2];
    const float* W2 = (const float*)d_in[3];
    const float* b2 = (const float*)d_in[4];
    float* out = (float*)d_out;

    char* ws = (char*)d_ws;
    bf16*  xbf   = (bf16*)(ws + O_XBF);
    bf16*  w1t   = (bf16*)(ws + O_W1T);
    bf16*  w2t   = (bf16*)(ws + O_W2T);
    float* headp = (float*)(ws + O_HEAD);
    float* tailp = (float*)(ws + O_TAIL);

    k_cvt_x<<<1536, 256, 0, stream>>>(x, xbf, NROW * HDIM);
    k_transpose_cvt<<<dim3(72, 48), 256, 0, stream>>>(W1, w1t, 1536, 2304);
    k_transpose_cvt<<<dim3(3, 72),  256, 0, stream>>>(W2, w2t, 2304, 96);
    k_gemm1<<<288, 256, 0, stream>>>(xbf, w1t, headp, tailp);
    k_pair<<<512, 256, 0, stream>>>(headp, tailp, b1, w2t, b2, out);
}
